// Gemma4Experts_20641612824866
// MI455X (gfx1250) — matrix operand order repaired and checked by reading, not compiled, not run on hardware
//
#include <hip/hip_runtime.h>
#include <hip/hip_bf16.h>

// ---------------------------------------------------------------------------
// Grouped MoE FFN for MI455X (gfx1250, wave32, WMMA).
//   g = x @ w1[e]^T ; u = x @ w3[e]^T ; h = gelu_tanh(g)*u ; out = h @ w2[e]^T
// bf16 matmuls (matching the torch/bf16 reference path), fp32 accumulate.
// BK=64 (32 WMMA per barrier pair), packed v_cvt_pk_bf16_f32 staging,
// hardware v_tanh_f32 epilogue, async global->LDS staging for the bf16 h tile.
// ---------------------------------------------------------------------------

typedef __attribute__((ext_vector_type(16))) __bf16 v16bf;
typedef __attribute__((ext_vector_type(8)))  float  v8f;

#define BM 128
#define BN 128
#define BK 64
#define NTHREADS 256   // 8 waves (wave32)

#if __has_builtin(__builtin_amdgcn_global_load_async_to_lds_b128)
#define USE_ASYNC_LDS 1
typedef int v4i_b128 __attribute__((vector_size(16)));
typedef __attribute__((address_space(1))) v4i_b128* gptr_b128;
typedef __attribute__((address_space(3))) v4i_b128* lptr_b128;
#else
#define USE_ASYNC_LDS 0
#endif

union Frag16 {
    v16bf v;
    uint4 q[2];
};

__device__ __forceinline__ unsigned short f2bf(float f) {
    unsigned u = __float_as_uint(f);
    unsigned r = u + 0x7FFFu + ((u >> 16) & 1u);   // round-to-nearest-even
    return (unsigned short)(r >> 16);
}

// Pack two floats into two bf16 (packed hardware convert when available).
__device__ __forceinline__ unsigned pack2bf(float lo, float hi) {
#if __has_builtin(__builtin_amdgcn_cvt_pk_bf16_f32)
    typedef __attribute__((ext_vector_type(2))) __bf16 v2bf;
    union { v2bf v; unsigned u; } c;
    c.v = __builtin_amdgcn_cvt_pk_bf16_f32(lo, hi);
    return c.u;
#else
    return (unsigned)f2bf(lo) | ((unsigned)f2bf(hi) << 16);
#endif
}

__device__ __forceinline__ float fast_tanh(float x) {
#if __has_builtin(__builtin_amdgcn_tanhf)
    return __builtin_amdgcn_tanhf(x);           // v_tanh_f32 (TRANS op)
#else
    return tanhf(x);
#endif
}

__device__ __forceinline__ float gelu_tanh(float x) {
    const float c0 = 0.7978845608028654f;  // sqrt(2/pi)
    const float c1 = 0.044715f;
    float x3 = x * x * x;
    return 0.5f * x * (1.0f + fast_tanh(c0 * (x + c1 * x3)));
}

// Which expert owns global token row `row` (groups are contiguous, sum to N).
__device__ __forceinline__ int expert_of(const int* nt, int E, int row) {
    int e = E - 1;
    int acc = 0;
    for (int i = 0; i < E; ++i) {
        acc += nt[i];
        if (row < acc) { e = i; break; }
    }
    return e;
}

// Stage a [rows x BK] fp32 tile (row stride ldg) into LDS as bf16.
__device__ __forceinline__ void load_tile_f32(const float* __restrict__ g, int ldg,
                                              unsigned short* lds, int rows) {
    const int nquads = rows * (BK / 4);            // float4 count
    for (int idx = threadIdx.x; idx < nquads; idx += NTHREADS) {
        int row = idx >> 4;                        // BK/4 = 16 quads per row
        int c4  = (idx & 15) * 4;
        const float4 v = *(const float4*)(g + (size_t)row * ldg + c4);
        uint2 p;
        p.x = pack2bf(v.x, v.y);
        p.y = pack2bf(v.z, v.w);
        *(uint2*)(lds + row * BK + c4) = p;
    }
}

// Stage a [rows x BK] bf16 tile into LDS; async direct-to-LDS when available.
__device__ __forceinline__ void load_tile_bf16(const unsigned short* __restrict__ g, int ldg,
                                               unsigned short* lds, int rows) {
    const int noct = rows * (BK / 8);              // 16-byte chunks
    for (int idx = threadIdx.x; idx < noct; idx += NTHREADS) {
        int row = idx >> 3;                        // BK/8 = 8 chunks per row
        int c8  = (idx & 7) * 8;
#if USE_ASYNC_LDS
        __builtin_amdgcn_global_load_async_to_lds_b128(
            (gptr_b128)(g + (size_t)row * ldg + c8),
            (lptr_b128)(lds + row * BK + c8),
            0, 0);
#else
        uint4 v = *(const uint4*)(g + (size_t)row * ldg + c8);
        *(uint4*)(lds + row * BK + c8) = v;
#endif
    }
}

__device__ __forceinline__ void wait_async_lds() {
#if USE_ASYNC_LDS
#if __has_builtin(__builtin_amdgcn_s_wait_asynccnt)
    __builtin_amdgcn_s_wait_asynccnt(0);
#else
    asm volatile("s_wait_asynccnt 0x0" ::: "memory");
#endif
#endif
}

// A fragment (16x32 bf16, M x K) from K-slice ks of a [.. x BK] LDS tile:
// lanes 0-15 row M=lane K{0..7,16..23}; lanes 16-31 row M=lane-16 K{8..15,24..31}.
__device__ __forceinline__ v16bf load_fragA(const unsigned short* lds, int rowBase,
                                            int ks, int lane) {
    int r  = rowBase + (lane & 15);
    int kh = (lane >> 4) * 8;
    const unsigned short* p = lds + r * BK + ks * 32;
    Frag16 f;
    f.q[0] = *(const uint4*)(p + kh);
    f.q[1] = *(const uint4*)(p + 16 + kh);
    return f.v;
}

// B fragment (32x16 bf16, K x N), weight rows are the N columns:
// lanes 0-15 col N=lane K{0..7,16..23}; lanes 16-31 col N=lane-16 K{8..15,24..31}.
__device__ __forceinline__ v16bf load_fragB(const unsigned short* lds, int colBase,
                                            int ks, int lane) {
    int n  = colBase + (lane & 15);
    int k0 = (lane >> 4) * 8;
    const unsigned short* p = lds + n * BK + ks * 32 + k0;
    Frag16 f;
    f.q[0] = *(const uint4*)p;
    f.q[1] = *(const uint4*)(p + 16);
    return f.v;
}

#define WMMA_BF16(a, b, c) \
    __builtin_amdgcn_wmma_f32_16x16x32_bf16(false, (a), false, (b), (short)0, (c), false, false)

// ---------------------------------------------------------------------------
// Kernel 1: h = gelu_tanh(x @ w1[e]^T) * (x @ w3[e]^T), h stored bf16 [N,H].
// grid = (N/BM, H/BN), block = 256.
// ---------------------------------------------------------------------------
__global__ __launch_bounds__(NTHREADS)
void moe_gate_up_kernel(const float* __restrict__ x,
                        const float* __restrict__ w1,
                        const float* __restrict__ w3,
                        const int*   __restrict__ nt,
                        unsigned short* __restrict__ hbuf,
                        int N, int D, int H, int E) {
    __shared__ unsigned short ldsX[BM * BK];
    __shared__ unsigned short ldsW1[BN * BK];
    __shared__ unsigned short ldsW3[BN * BK];

    const int rowStart = blockIdx.x * BM;          // global token rows
    const int colStart = blockIdx.y * BN;          // H columns
    const int e = expert_of(nt, E, rowStart);

    const float* xP  = x  + (size_t)rowStart * D;
    const float* w1P = w1 + (size_t)e * H * D + (size_t)colStart * D;
    const float* w3P = w3 + (size_t)e * H * D + (size_t)colStart * D;

    const int lane  = threadIdx.x & 31;
    const int wid   = threadIdx.x >> 5;
    const int waveM = (wid >> 2) * 64;             // 2 wave-rows x 4 wave-cols
    const int waveN = (wid & 3) * 32;

    v8f gAcc[4][2], uAcc[4][2];
    #pragma unroll
    for (int i = 0; i < 4; ++i)
        #pragma unroll
        for (int j = 0; j < 2; ++j) { gAcc[i][j] = {}; uAcc[i][j] = {}; }

    const int ksteps = D / BK;
    for (int kt = 0; kt < ksteps; ++kt) {
        load_tile_f32(xP  + kt * BK, D, ldsX,  BM);
        load_tile_f32(w1P + kt * BK, D, ldsW1, BN);
        load_tile_f32(w3P + kt * BK, D, ldsW3, BN);
        if (kt + 1 < ksteps) {                      // global_prefetch_b8 next K-tile
            __builtin_prefetch(xP  + (kt + 1) * BK + (size_t)lane * D, 0, 3);
            __builtin_prefetch(w1P + (kt + 1) * BK + (size_t)lane * D, 0, 3);
            __builtin_prefetch(w3P + (kt + 1) * BK + (size_t)lane * D, 0, 3);
        }
        __syncthreads();

        #pragma unroll
        for (int ks = 0; ks < BK / 32; ++ks) {
            v16bf a[4];
            #pragma unroll
            for (int ms = 0; ms < 4; ++ms)
                a[ms] = load_fragA(ldsX, waveM + ms * 16, ks, lane);

            #pragma unroll
            for (int ns = 0; ns < 2; ++ns) {
                v16bf b1 = load_fragB(ldsW1, waveN + ns * 16, ks, lane);
                v16bf b3 = load_fragB(ldsW3, waveN + ns * 16, ks, lane);
                #pragma unroll
                for (int ms = 0; ms < 4; ++ms) {
                    gAcc[ms][ns] = WMMA_BF16(a[ms], b1, gAcc[ms][ns]);
                    uAcc[ms][ns] = WMMA_BF16(a[ms], b3, uAcc[ms][ns]);
                }
            }
        }
        __syncthreads();
    }

    // Epilogue: h = gelu(g) * u -> bf16. C layout: elem i -> M = i + (lane>=16)*8.
    const int l15 = lane & 15;
    const int lh8 = (lane >> 4) * 8;
    #pragma unroll
    for (int ms = 0; ms < 4; ++ms) {
        #pragma unroll
        for (int ns = 0; ns < 2; ++ns) {
            int col = colStart + waveN + ns * 16 + l15;
            #pragma unroll
            for (int i = 0; i < 8; ++i) {
                int row = rowStart + waveM + ms * 16 + lh8 + i;
                float g = gAcc[ms][ns][i];
                float u = uAcc[ms][ns][i];
                hbuf[(size_t)row * H + col] = f2bf(gelu_tanh(g) * u);
            }
        }
    }
}

// ---------------------------------------------------------------------------
// Kernel 2: out = h @ w2[e]^T, fp32 out [N,D]. grid = (N/BM, D/BN), block=256.
// ---------------------------------------------------------------------------
__global__ __launch_bounds__(NTHREADS)
void moe_down_kernel(const unsigned short* __restrict__ hbuf,
                     const float* __restrict__ w2,
                     const int*   __restrict__ nt,
                     float* __restrict__ out,
                     int N, int D, int H, int E) {
    __shared__ unsigned short ldsH[BM * BK];
    __shared__ unsigned short ldsW[BN * BK];

    const int rowStart = blockIdx.x * BM;
    const int colStart = blockIdx.y * BN;          // D columns
    const int e = expert_of(nt, E, rowStart);

    const unsigned short* hP = hbuf + (size_t)rowStart * H;
    const float* wP = w2 + (size_t)e * D * H + (size_t)colStart * H;

    const int lane  = threadIdx.x & 31;
    const int wid   = threadIdx.x >> 5;
    const int waveM = (wid >> 2) * 64;
    const int waveN = (wid & 3) * 32;

    v8f acc[4][2];
    #pragma unroll
    for (int i = 0; i < 4; ++i)
        #pragma unroll
        for (int j = 0; j < 2; ++j) acc[i][j] = {};

    const int ksteps = H / BK;
    for (int kt = 0; kt < ksteps; ++kt) {
        load_tile_bf16(hP + kt * BK, H, ldsH, BM);   // async direct-to-LDS
        load_tile_f32 (wP + kt * BK, H, ldsW, BN);
        if (kt + 1 < ksteps) {
            __builtin_prefetch(hP + (kt + 1) * BK + (size_t)lane * H, 0, 3);
            __builtin_prefetch(wP + (kt + 1) * BK + (size_t)lane * H, 0, 3);
        }
        wait_async_lds();
        __syncthreads();

        #pragma unroll
        for (int ks = 0; ks < BK / 32; ++ks) {
            v16bf a[4];
            #pragma unroll
            for (int ms = 0; ms < 4; ++ms)
                a[ms] = load_fragA(ldsH, waveM + ms * 16, ks, lane);

            #pragma unroll
            for (int ns = 0; ns < 2; ++ns) {
                v16bf b = load_fragB(ldsW, waveN + ns * 16, ks, lane);
                #pragma unroll
                for (int ms = 0; ms < 4; ++ms)
                    acc[ms][ns] = WMMA_BF16(a[ms], b, acc[ms][ns]);
            }
        }
        __syncthreads();
    }

    const int l15 = lane & 15;
    const int lh8 = (lane >> 4) * 8;
    #pragma unroll
    for (int ms = 0; ms < 4; ++ms) {
        #pragma unroll
        for (int ns = 0; ns < 2; ++ns) {
            int col = colStart + waveN + ns * 16 + l15;
            #pragma unroll
            for (int i = 0; i < 8; ++i) {
                int row = rowStart + waveM + ms * 16 + lh8 + i;
                out[(size_t)row * D + col] = acc[ms][ns][i];
            }
        }
    }
}

// ---------------------------------------------------------------------------
// Launch
// ---------------------------------------------------------------------------
extern "C" void kernel_launch(void* const* d_in, const int* in_sizes, int n_in,
                              void* d_out, int out_size, void* d_ws, size_t ws_size,
                              hipStream_t stream) {
    const int E = 8, D = 1024, H = 4096, N = 8192;
    (void)in_sizes; (void)n_in; (void)out_size; (void)ws_size;

    const float* x  = (const float*)d_in[0];
    const float* w1 = (const float*)d_in[1];
    const float* w2 = (const float*)d_in[2];
    const float* w3 = (const float*)d_in[3];
    const int*   nt = (const int*)d_in[4];
    float* out = (float*)d_out;

    unsigned short* hbuf = (unsigned short*)d_ws;   // N*H bf16 = 64 MB scratch

    dim3 gridA(N / BM, H / BN);                     // 64 x 32
    moe_gate_up_kernel<<<gridA, NTHREADS, 0, stream>>>(x, w1, w3, nt, hbuf, N, D, H, E);

    dim3 gridB(N / BM, D / BN);                     // 64 x 8
    moe_down_kernel<<<gridB, NTHREADS, 0, stream>>>(hbuf, w2, nt, out, N, D, H, E);
}
